// Biaffine_54623394070940
// MI455X (gfx1250) — compile-verified, matrix operand order checked by reading, not run on hardware
//
#include <hip/hip_runtime.h>
#include <hip/hip_bf16.h>

// Biaffine: out[b,x,y,o] = sum_ij in1[b,x,i]*w1[i,o,j]*in2[b,y,j]
//                        + in1[b,x,:]@wA[:,o] + in1[b,y,:]@wB[:,o] + bias[o]
// B=4, S=512, IN=512, OUT=112. ~240 GFLOP -> compute-bound on bf16 WMMA;
// operands (~63 MB bf16) are L2-resident, output (470 MB) streams out.

#define BB   4
#define SS   512
#define INN  512
#define OUTT 112

typedef __attribute__((ext_vector_type(16))) __bf16 v16bf;
typedef __attribute__((ext_vector_type(8)))  __bf16 v8bf;
typedef __attribute__((ext_vector_type(8)))  float  v8f;
typedef __attribute__((ext_vector_type(4)))  unsigned int v4u;
typedef __attribute__((ext_vector_type(8)))  int v8i;
typedef __attribute__((ext_vector_type(4)))  int v4i;

// LDS row strides (bf16 elements), padded for conflict-free b128 fragment loads.
// TDM reproduces this padding via pad_interval/pad_amount.
#define A_STR 520   // 512 el (1024 B) + 16 B pad
#define B_STR 40    //  32 el (  64 B) + 16 B pad

// B ring buffer base (LDS), computed arithmetically — no pointer arrays of
// addrspace(3) pointers (ld.lld cannot statically initialize those).
#define BBUF(i) (smem + 128 * A_STR + (size_t)(i) * (128 * B_STR))

#if __has_builtin(__builtin_amdgcn_tensor_load_to_lds) && __has_builtin(__builtin_amdgcn_s_wait_tensorcnt)
#define HAVE_TDM 1
#else
#define HAVE_TDM 0
#endif

__device__ __forceinline__ unsigned ldsaddr(const void* p) {
    return (unsigned)(unsigned long long)p;     // low 32 bits of generic LDS addr = byte offset
}

#if HAVE_TDM
// Issue a 2-D TENSOR_LOAD_TO_LDS: tile_d1 rows of tile_d0 bf16 elements,
// global row stride = stride_el elements, LDS rows padded per pad_* encoding.
__device__ __forceinline__ void tdm_load_2d(unsigned lds_addr, const void* gptr,
                                            unsigned tile_d0, unsigned tile_d1,
                                            unsigned stride_el,
                                            unsigned pad_interval, unsigned pad_amount)
{
    unsigned long long ga = (unsigned long long)gptr;
    v4u g0;
    g0[0] = 1u;                                               // count=1, user descriptor
    g0[1] = lds_addr;                                         // D#.lds_addr
    g0[2] = (unsigned)ga;                                     // global_addr[31:0]
    g0[3] = (unsigned)((ga >> 32) & 0x01FFFFFFull) | (2u << 30); // addr[56:32] | type=2
    const unsigned td0 = 0x40000000u, td1 = 0x40000000u;      // huge dims -> no OOB clip
    v8i g1;
    g1[0] = (int)((1u << 16)                                  // data_size = 2 bytes
                | (1u << 20)                                  // pad_enable
                | (pad_interval << 22) | (pad_amount << 25));
    g1[1] = (int)((td0 & 0xFFFFu) << 16);                     // tensor_dim0[15:0]
    g1[2] = (int)((td0 >> 16) | ((td1 & 0xFFFFu) << 16));     // td0[31:16] | td1[15:0]
    g1[3] = (int)((td1 >> 16) | (tile_d0 << 16));             // td1[31:16] | tile_dim0
    g1[4] = (int)(tile_d1 & 0xFFFFu);                         // tile_dim1 (tile_dim2=0)
    g1[5] = (int)stride_el;                                   // tensor_dim0_stride[31:0]
    g1[6] = 0;
    g1[7] = 0;
    v4i z4 = {0, 0, 0, 0};                                    // 2-D: groups 2/3 unused
#if __clang_major__ >= 23
    v8i z8 = {0, 0, 0, 0, 0, 0, 0, 0};
    __builtin_amdgcn_tensor_load_to_lds(g0, g1, z4, z4, z8, 0);
#else
    __builtin_amdgcn_tensor_load_to_lds(g0, g1, z4, z4, 0);
#endif
}
#endif

// Fragment load: 8 bf16 at p, 8 bf16 at p+16 (CDNA5 16-bit A/B operand layout:
// lanes 0-15 hold K {0..7,16..23}; lanes 16-31 hold K {8..15,24..31}).
__device__ __forceinline__ v16bf load_frag(const __bf16* p) {
    v8bf lo = *reinterpret_cast<const v8bf*>(p);
    v8bf hi = *reinterpret_cast<const v8bf*>(p + 16);
    v16bf r;
#pragma unroll
    for (int e = 0; e < 8; ++e) { r[e] = lo[e]; r[e + 8] = hi[e]; }
    return r;
}

// ---------------------------------------------------------------------------
// Prep 1: f32 -> bf16 copies of input1/input2 (coalesced)
// ---------------------------------------------------------------------------
__global__ __launch_bounds__(256)
void convert_inputs(const float* __restrict__ in1, const float* __restrict__ in2,
                    __bf16* __restrict__ in1b, __bf16* __restrict__ in2b)
{
    const size_t base = (size_t)blockIdx.x * 1024;
#pragma unroll
    for (int k = 0; k < 4; ++k) {
        size_t i = base + threadIdx.x + (size_t)k * 256;   // covers B*S*IN = 1M
        in1b[i] = (__bf16)in1[i];
        in2b[i] = (__bf16)in2[i];
    }
}

// ---------------------------------------------------------------------------
// Prep 2: w1t[o][j][i] = (bf16) w1[i][o][j]   (32x32 LDS tile transpose)
// ---------------------------------------------------------------------------
__global__ __launch_bounds__(256)
void transpose_w1(const float* __restrict__ w1, __bf16* __restrict__ w1t)
{
    __shared__ float tile[32][33];
    const int j0 = blockIdx.x * 32, i0 = blockIdx.y * 32, o = blockIdx.z;
    const int c = threadIdx.x & 31, seg = threadIdx.x >> 5;   // 8 segs x 4 rows
#pragma unroll
    for (int rr = 0; rr < 4; ++rr) {
        int il = seg * 4 + rr;
        tile[il][c] = w1[(size_t)(i0 + il) * (OUTT * INN) + (size_t)o * INN + (j0 + c)];
    }
    __syncthreads();
#pragma unroll
    for (int rr = 0; rr < 4; ++rr) {
        int jl = seg * 4 + rr;
        w1t[(size_t)o * INN * INN + (size_t)(j0 + jl) * INN + (i0 + c)] =
            (__bf16)tile[c][jl];
    }
}

// ---------------------------------------------------------------------------
// Prep 3: termA/termB rank-1 corrections (tiny, VALU)
// ---------------------------------------------------------------------------
__global__ __launch_bounds__(128)
void biaffine_terms(const float* __restrict__ in1, const float* __restrict__ w2,
                    float* __restrict__ termA, float* __restrict__ termB)
{
    const int bs = blockIdx.x;
    const int o  = threadIdx.x;
    if (o >= OUTT) return;
    const float* row = in1 + (size_t)bs * INN;
    float a = 0.f, c = 0.f;
    for (int i = 0; i < INN; ++i) {
        const float v = row[i];
        a += v * w2[(size_t)i * OUTT + o];
        c += v * w2[(size_t)(i + INN) * OUTT + o];
    }
    termA[(size_t)bs * OUTT + o] = a;
    termB[(size_t)bs * OUTT + o] = c;
}

// One 64x128(N) x K=512 GEMM pass: acc += Alds @ B, where B tiles (128x32 bf16)
// stream from `tbase` (row stride 512 el) through 3 LDS buffers via TDM,
// depth-2 prefetch, ONE barrier per 32-deep K chunk.
#define GEMM_PASS(Alds, tbase)                                                      \
    {                                                                               \
        _Pragma("unroll 1")                                                         \
        for (int kc = 0; kc < 16; ++kc) {                                           \
            const __bf16* bbuf = BBUF(kc % 3);                                      \
            const int kb = kc * 32;                                                 \
            (void)kb;                                                               \
            BARRIER_PREFETCH(tbase, kc, bbuf)                                       \
            v16bf af = load_frag(&(Alds)[(wx + arow) * A_STR + kb + khalf]);        \
            v16bf bfr[4];                                                           \
            _Pragma("unroll")                                                       \
            for (int f = 0; f < 4; ++f)                                             \
                bfr[f] = load_frag(&bbuf[(wn + f * 16 + arow) * B_STR + khalf]);    \
            _Pragma("unroll")                                                       \
            for (int f = 0; f < 4; ++f)                                             \
                acc[f] = __builtin_amdgcn_wmma_f32_16x16x32_bf16(                   \
                    false, af, false, bfr[f], (short)0, acc[f], false, false);      \
        }                                                                           \
        __syncthreads();  /* protect ring slots before next pass preloads */        \
    }

#if HAVE_TDM
// wait for chunk kc; publish; then (wave0) prefetch chunk kc+2
#define BARRIER_PREFETCH(tbase, kc, bbuf)                                           \
    if (wave == 0) {                                                                \
        if (kc < 15) __builtin_amdgcn_s_wait_tensorcnt(1);                          \
        else         __builtin_amdgcn_s_wait_tensorcnt(0);                          \
    }                                                                               \
    __syncthreads();                                                                \
    if (wave == 0 && kc < 14)                                                       \
        tdm_load_2d(ldsaddr(BBUF((kc + 2) % 3)), (tbase) + (kc + 2) * 32,           \
                    32, 128, 512, 3, 3);
#define PRELOAD(tbase)                                                              \
    if (wave == 0) {                                                                \
        tdm_load_2d(ldsaddr(BBUF(0)), (tbase),      32, 128, 512, 3, 3);            \
        tdm_load_2d(ldsaddr(BBUF(1)), (tbase) + 32, 32, 128, 512, 3, 3);            \
    }
#else
// fallback: synchronous bf16 staging, 32 B per thread, two barriers per chunk
#define BARRIER_PREFETCH(tbase, kc, bbuf)                                           \
    __syncthreads();                                                                \
    {                                                                               \
        int jl = tid >> 1, half = (tid & 1) * 16;                                   \
        const v8bf* s = reinterpret_cast<const v8bf*>((tbase) + (size_t)jl * INN + kb + half); \
        v8bf t0 = s[0], t1 = s[1];                                                  \
        v8bf* d = reinterpret_cast<v8bf*>(const_cast<__bf16*>(&bbuf[jl * B_STR + half])); \
        d[0] = t0; d[1] = t1;                                                       \
    }                                                                               \
    __syncthreads();
#define PRELOAD(tbase)
#endif

// ---------------------------------------------------------------------------
// Main: fused two-stage WMMA per (b, o, 64-row x-tile)
//   stage 1: T(64x512)  = in1_tile @ w1t[o]      (T -> LDS bf16)
//   stage 2: out(64x512)= T @ in2^T  + termA + termB + bias
// ---------------------------------------------------------------------------
__global__ __launch_bounds__(256)
void biaffine_main(const __bf16* __restrict__ in1b,
                   const __bf16* __restrict__ in2b,
                   const __bf16* __restrict__ w1t,
                   const float* __restrict__ w2,
                   const float* __restrict__ termA,
                   const float* __restrict__ termB,
                   float* __restrict__ out)
{
    extern __shared__ __bf16 smem[];
    __bf16* A1 = smem;                              //  64 x A_STR
    __bf16* T  = smem + 64 * A_STR;                 //  64 x A_STR
    // B ring: 3 x (128 x B_STR) starting at smem + 128*A_STR, via BBUF(i)

    const int xt = blockIdx.x, o = blockIdx.y, b = blockIdx.z;
    const int x0 = xt * 64;

    const int tid   = threadIdx.x;
    const int lane  = tid & 31;
    const int wave  = tid >> 5;                 // 0..7
    const int wx    = (wave & 3) * 16;          // wave row block within 64
    const int wn    = (wave >> 2) * 64;         // wave col block within 128
    const int arow  = lane & 15;
    const int khalf = (lane >> 4) * 8;
    const int rbase = wx + ((lane >> 4) * 8);

    const __bf16* a_src = in1b + ((size_t)b * SS + x0) * INN;
    const __bf16* wbase = w1t + (size_t)o * INN * INN;
    const __bf16* ybase = in2b + (size_t)b * SS * INN;

    // ---- A tile: 64 x 512 bf16 into padded LDS ----
#if HAVE_TDM
    if (wave == 0) {
        tdm_load_2d(ldsaddr(A1), a_src, 512, 64, 512, /*pad@1024B*/7, /*16B*/3);
        __builtin_amdgcn_s_wait_tensorcnt(0);
    }
#else
#pragma unroll
    for (int it = 0; it < 16; ++it) {
        int c16 = tid + it * 256;               // 4096 16-byte chunks
        int r = c16 >> 6, cc = c16 & 63;
        *reinterpret_cast<v8bf*>(&A1[r * A_STR + cc * 8]) =
            *reinterpret_cast<const v8bf*>(&a_src[(size_t)r * INN + cc * 8]);
    }
    __syncthreads();
#endif

    // ================= stage 1: T = A1 @ W_o =================
    for (int nt = 0; nt < 4; ++nt) {
        v8f acc[4] = {};
        const __bf16* tbase = wbase + (size_t)(nt * 128) * INN;
        PRELOAD(tbase)
        GEMM_PASS(A1, tbase)
        // spill T tile to LDS as bf16 (C/D layout: lane<16 -> M=r, else M=r+8)
#pragma unroll
        for (int f = 0; f < 4; ++f) {
            const int n = nt * 128 + wn + f * 16 + arow;
#pragma unroll
            for (int r = 0; r < 8; ++r)
                T[(rbase + r) * A_STR + n] = (__bf16)acc[f][r];
        }
    }

    // ================= stage 2: out = T @ in2^T + logits2 =================
    const float biasv = w2[(size_t)(2 * INN) * OUTT + o];
    for (int yt = 0; yt < 4; ++yt) {
        v8f acc[4] = {};
        const __bf16* tbase = ybase + (size_t)(yt * 128) * INN;
        PRELOAD(tbase)
        GEMM_PASS(T, tbase)
        // epilogue: + termA[x,o] + termB[y,o] + bias[o]
        float tA[8];
#pragma unroll
        for (int r = 0; r < 8; ++r)
            tA[r] = termA[((size_t)b * SS + (x0 + rbase + r)) * OUTT + o];
#pragma unroll
        for (int f = 0; f < 4; ++f) {
            const int y  = yt * 128 + wn + f * 16 + arow;
            const float tB = termB[((size_t)b * SS + y) * OUTT + o] + biasv;
#pragma unroll
            for (int r = 0; r < 8; ++r) {
                const int x = x0 + rbase + r;
                out[(((size_t)b * SS + x) * SS + y) * OUTT + o] = acc[f][r] + tA[r] + tB;
            }
        }
    }
}

// ---------------------------------------------------------------------------
extern "C" void kernel_launch(void* const* d_in, const int* in_sizes, int n_in,
                              void* d_out, int out_size, void* d_ws, size_t ws_size,
                              hipStream_t stream)
{
    const float* in1 = (const float*)d_in[0];   // (B,S,IN)
    const float* in2 = (const float*)d_in[1];   // (B,S,IN)
    const float* w1  = (const float*)d_in[2];   // (IN,OUT,IN)
    const float* w2  = (const float*)d_in[3];   // (2*IN+1,OUT)
    float* out = (float*)d_out;                 // (B,S,S,OUT)

    // Workspace layout (total ~64.8 MB):
    //   w1t  bf16 [OUT][IN][IN]   58,720,256 B
    //   in1b bf16 [B][S][IN]       2,097,152 B
    //   in2b bf16 [B][S][IN]       2,097,152 B
    //   termA/termB f32 [B][S][OUT]  917,504 B each
    char* ws = (char*)d_ws;
    __bf16* w1t  = (__bf16*)ws;
    __bf16* in1b = (__bf16*)(ws + 58720256);
    __bf16* in2b = (__bf16*)(ws + 58720256 + 2097152);
    float*  termA = (float*)(ws + 58720256 + 2 * 2097152);
    float*  termB = termA + (size_t)BB * SS * OUTT;

    convert_inputs<<<1024, 256, 0, stream>>>(in1, in2, in1b, in2b);
    transpose_w1<<<dim3(16, 16, OUTT), 256, 0, stream>>>(w1, w1t);
    biaffine_terms<<<BB * SS, 128, 0, stream>>>(in1, w2, termA, termB);

    dim3 grid(SS / 64, OUTT, BB);
    size_t lds = (size_t)(128 * A_STR + 3 * 128 * B_STR) * sizeof(__bf16); // 163,840 B
    biaffine_main<<<grid, 256, lds, stream>>>(in1b, in2b, w1t, w2, termA, termB, out);
}
